// SO3Conv_13417477832859
// MI455X (gfx1250) — matrix operand order repaired and checked by reading, not compiled, not run on hardware
//
#include <hip/hip_runtime.h>
#include <stdint.h>

// Problem constants (match reference)
#define BATCH   512
#define FIN     64
#define FOUT    64
#define LMAX    8
#define NROT    216
#define TOTAL   969

// Workspace layout (bf16 elements):
//   Abf: per-l row-major (M_l x K_l), M_l=512d, K_l=64d  -> 32768*969 elems (63.5 MB)
//   Bt : per-l row-major (N_l x K_l) (B transposed)      ->  4096*969 elems ( 7.9 MB)
#define A_ELEMS 31752192ull   // 32768*969
#define B_ELEMS  3969024ull   //  4096*969

typedef __attribute__((ext_vector_type(16))) __bf16 v16bf;
typedef __attribute__((ext_vector_type(8)))  float  v8f;
typedef __attribute__((ext_vector_type(4)))  unsigned int u32x4;

static __device__ __forceinline__ unsigned short f32_bf16_rne(float f) {
    unsigned u = __float_as_uint(f);
    u += 0x7FFFu + ((u >> 16) & 1u);
    return (unsigned short)(u >> 16);
}

// off(l) = sum_{l'<l} (2l'+1)^2 = l(2l-1)(2l+1)/3
static __device__ __forceinline__ int seg_off(int l) {
    return l * (2 * l - 1) * (2 * l + 1) / 3;
}

// magic = ceil(2^22/d): (v*magic)>>22 == v/d exactly for v < 2^22/16 (d<=17)
static __device__ __forceinline__ unsigned magic_of(int d) {
    return (0x400000u + (unsigned)d - 1u) / (unsigned)d;
}
static __device__ __forceinline__ unsigned fdiv(unsigned v, unsigned m) {
    return (unsigned)(((unsigned long long)v * m) >> 22);
}

// ---------------------------------------------------------------------------
// Repack X -> Abf (bf16):  Abf_l[(b*d+i)*K_l + (x*d+u)] = x[b,x,off+u*d+i]
// One block per (b,x,l); coalesced reads, one-time scatter writes (L2 absorbs).
// ---------------------------------------------------------------------------
__global__ __launch_bounds__(256)
void repack_x_kernel(const float* __restrict__ X, unsigned short* __restrict__ Abf) {
    const int l   = blockIdx.z;
    const int d   = 2 * l + 1;
    const int off = seg_off(l);
    const int e   = blockIdx.x;            // b*64 + x
    const int b   = e >> 6, xc = e & 63;
    const int Kl  = FIN * d;
    const unsigned md = magic_of(d);
    const float* src = X + (size_t)e * TOTAL + off;
    unsigned short* dst = Abf + (size_t)32768 * off;
    const int dd = d * d;
    for (int t = threadIdx.x; t < dd; t += 256) {
        const unsigned u = fdiv((unsigned)t, md);
        const unsigned ii = (unsigned)t - u * d;       // t = u*d + ii
        dst[((size_t)b * d + ii) * Kl + xc * d + u] = f32_bf16_rne(src[t]);
    }
}

// ---------------------------------------------------------------------------
// psi + repack -> Bt (bf16), with alpha = 1/sqrt(64d) folded in:
//   Bt_l[(y*d+w)*K_l + (x*d+u)] = alpha_l * psi[x,y,off+u*d+w]
// psi[x,y,i] = (1/sqrt(216)) * sum_n w[x,y,n]*D[n,i]; D stays hot in L2.
// ---------------------------------------------------------------------------
__global__ __launch_bounds__(256)
void psi_kernel(const float* __restrict__ w, const float* __restrict__ D,
                unsigned short* __restrict__ Bt) {
    const int xy = blockIdx.x;             // x*64 + y
    const int xc = xy >> 6, y = xy & 63;
    __shared__ float wl[NROT];
    for (int n = threadIdx.x; n < NROT; n += 256) wl[n] = w[xy * NROT + n];
    __syncthreads();
    const float s = 0.06804138174397717f;  // 1/sqrt(216)
    for (int i = threadIdx.x; i < TOTAL; i += 256) {
        float acc = 0.0f;
        #pragma unroll 8
        for (int n = 0; n < NROT; ++n)
            acc = fmaf(wl[n], D[n * TOTAL + i], acc);
        int l = 0;
        while (l < LMAX && i >= seg_off(l + 1)) ++l;
        const int d = 2 * l + 1, off = seg_off(l);
        const unsigned md = magic_of(d);
        const unsigned r = (unsigned)(i - off);
        const unsigned u = fdiv(r, md), wc = r - u * d;      // i = off + u*d + wc
        const float alpha = 0.125f * rsqrtf((float)d);       // 1/sqrt(64d)
        Bt[(size_t)4096 * off + ((size_t)y * d + wc) * (FIN * d) + xc * d + u] =
            f32_bf16_rne(acc * s * alpha);
    }
}

// ---------------------------------------------------------------------------
// Kernel 3: per-l GEMM on packed bf16.  out[(b,i),(y,w)] = A x Bt^T (alpha folded)
// BM=128 BN=64 BK=64; 4 waves, wave tile 32x64 => 2x4 WMMA x 2 k-steps/stage.
// Tiles staged with double-buffered global_load_async_to_lds_b128 (ASYNCcnt).
// ---------------------------------------------------------------------------
#define BM 128
#define BN 64
#define BK 64
#define LDA 72   // bf16 elems per LDS row (144B: 16B-aligned, staggers banks)
#define LDB 72

__global__ __launch_bounds__(128)
void so3_wmma_kernel(const unsigned short* __restrict__ Abf,
                     const unsigned short* __restrict__ Bt,
                     float* __restrict__ out) {
    const int l   = blockIdx.z;
    const int d   = 2 * l + 1;
    const int off = seg_off(l);
    const int Ml  = BATCH * d;
    const int Nl  = FOUT  * d;
    const int Kl  = FIN   * d;
    const int rowBase = blockIdx.x * BM;
    const int colBase = blockIdx.y * BN;
    if (rowBase >= Ml || colBase >= Nl) return;

    __shared__ __align__(16) unsigned short sA[2][BM * LDA];  // 2 x 18 KB
    __shared__ __align__(16) unsigned short sB[2][BN * LDB];  // 2 x  9 KB

    const int tid  = threadIdx.x;      // 0..127
    const int lane = tid & 31;
    const int wv   = tid >> 5;         // 0..3 -> wave tile rows m0..m0+31
    const int m0   = wv * 32;
    const int lh   = lane >> 4;
    const int ll   = lane & 15;

    const unsigned short* Ab = Abf + (size_t)32768 * off;
    const unsigned short* Bb = Bt  + (size_t)4096  * off;

    // ---- async-copy assignment: 16B chunks, 8 chunks per 64-elem row ----
    // A: 1024 chunks -> 8 issues/thread; B: 512 chunks -> 4 issues/thread.
    // chunk = tid + j*128  =>  row = (tid>>3) + 16*j,  col = (tid&7)*8 (fixed)
    const int r0 = tid >> 3;
    const int q0 = (tid & 7) * 8;
    const unsigned long long rowStep = (unsigned long long)(32 * Kl); // 16 rows, bytes

    unsigned long long gA = (unsigned long long)(uintptr_t)(Ab + (size_t)(rowBase + r0) * Kl + q0);
    unsigned long long gB = (unsigned long long)(uintptr_t)(Bb + (size_t)(colBase + r0) * Kl + q0);

    // LDS byte offsets (generic-pointer low 32 bits == LDS offset, ISA 10.2)
    const unsigned lA_0 = (unsigned)(uintptr_t)&sA[0][r0 * LDA + q0];
    const unsigned lA_1 = (unsigned)(uintptr_t)&sA[1][r0 * LDA + q0];
    const unsigned lB_0 = (unsigned)(uintptr_t)&sB[0][r0 * LDB + q0];
    const unsigned lB_1 = (unsigned)(uintptr_t)&sB[1][r0 * LDB + q0];
    const unsigned ldsRowStep = 16 * LDA * 2;   // == 16*LDB*2 == 2304 bytes

    auto issue_tile = [&](int s_) {
        const unsigned la = s_ ? lA_1 : lA_0;
        const unsigned lb = s_ ? lB_1 : lB_0;
        unsigned long long ga = gA, gb = gB;
        #pragma unroll
        for (int j = 0; j < 8; ++j) {
            asm volatile("global_load_async_to_lds_b128 %0, %1, off"
                         :: "v"(la + j * ldsRowStep), "v"(ga) : "memory");
            ga += rowStep;
        }
        #pragma unroll
        for (int j = 0; j < 4; ++j) {
            asm volatile("global_load_async_to_lds_b128 %0, %1, off"
                         :: "v"(lb + j * ldsRowStep), "v"(gb) : "memory");
            gb += rowStep;
        }
        gA += 2 * BK;   // advance one K stage (bytes)
        gB += 2 * BK;
    };

    v8f acc[2][4] = {};
    const int nk = Kl / BK;            // = d, exact

    issue_tile(0);
    for (int kt = 0; kt < nk; ++kt) {
        const int cur = kt & 1;
        if (kt + 1 < nk) {
            issue_tile((kt + 1) & 1);
            asm volatile("s_wait_asynccnt 0xc" ::: "memory");  // prev 12 done
        } else {
            asm volatile("s_wait_asynccnt 0x0" ::: "memory");
        }
        __syncthreads();

        // two WMMA k-steps per staged BK=64 tile
        #pragma unroll
        for (int ks = 0; ks < 2; ++ks) {
            union Frag { v16bf v; u32x4 q[2]; } a[2], bf[4];
            #pragma unroll
            for (int tm = 0; tm < 2; ++tm) {
                const unsigned short* p = &sA[cur][(m0 + tm * 16 + ll) * LDA + ks * 32];
                a[tm].q[0] = *(const u32x4*)(p + 8 * lh);       // K = 8*lh..+7
                a[tm].q[1] = *(const u32x4*)(p + 16 + 8 * lh);  // K = 16+8*lh..+7
            }
            #pragma unroll
            for (int tn = 0; tn < 4; ++tn) {
                const unsigned short* p = &sB[cur][(tn * 16 + ll) * LDB + ks * 32];
                bf[tn].q[0] = *(const u32x4*)(p + 8 * lh);      // K = 8*lh..+7
                bf[tn].q[1] = *(const u32x4*)(p + 16 + 8 * lh); // K = 16+8*lh..+7
            }
            #pragma unroll
            for (int tm = 0; tm < 2; ++tm)
                #pragma unroll
                for (int tn = 0; tn < 4; ++tn)
                    acc[tm][tn] = __builtin_amdgcn_wmma_f32_16x16x32_bf16(
                        false, a[tm].v, false, bf[tn].v,
                        (short)0, acc[tm][tn], false, false);
        }
        __syncthreads();
    }

    // ---- epilogue: alpha already folded; magic-divide scatter ----
    const unsigned md = magic_of(d);
    #pragma unroll
    for (int tm = 0; tm < 2; ++tm) {
        #pragma unroll
        for (int tn = 0; tn < 4; ++tn) {
            #pragma unroll
            for (int r = 0; r < 8; ++r) {
                const unsigned R = (unsigned)(rowBase + m0 + tm * 16 + r + 8 * lh);
                const unsigned c = (unsigned)(colBase + tn * 16 + ll);
                const unsigned bb = fdiv(R, md), ii = R - bb * d;
                const unsigned yy = fdiv(c, md), wc = c - yy * d;
                out[(size_t)bb * (FOUT * TOTAL) + (size_t)yy * TOTAL +
                    (size_t)(off + (int)(wc * (unsigned)d + ii))] = acc[tm][tn][r];
            }
        }
    }
}

// ---------------------------------------------------------------------------
extern "C" void kernel_launch(void* const* d_in, const int* in_sizes, int n_in,
                              void* d_out, int out_size, void* d_ws, size_t ws_size,
                              hipStream_t stream) {
    (void)in_sizes; (void)n_in; (void)out_size; (void)ws_size;
    const float* x = (const float*)d_in[0];   // (512, 64, 969)
    const float* w = (const float*)d_in[1];   // (64, 64, 216)
    const float* D = (const float*)d_in[2];   // (216, 969)
    float* out = (float*)d_out;               // (512, 64, 969)

    unsigned short* Abf = (unsigned short*)d_ws;        // 63.5 MB
    unsigned short* Bt  = Abf + A_ELEMS;                //  7.9 MB

    repack_x_kernel<<<dim3(BATCH * FIN, 1, LMAX + 1), dim3(256), 0, stream>>>(x, Abf);
    psi_kernel<<<dim3(FIN * FOUT), dim3(256), 0, stream>>>(w, D, Bt);

    // grid covers largest l (M=8704 -> 68 tiles, N=1088 -> 17 tiles);
    // out-of-range tiles for smaller l exit immediately.
    so3_wmma_kernel<<<dim3(68, 17, LMAX + 1), dim3(128), 0, stream>>>(Abf, Bt, out);
}
